// LSTM_66967130079764
// MI455X (gfx1250) — matrix operand order repaired and checked by reading, not compiled, not run on hardware
//
#include <hip/hip_runtime.h>
#include <hip/hip_bf16.h>

// ---------------------------------------------------------------------------
// Bidirectional 2-layer LSTM (T=512, B=256, H=128) + FC(256->6) for gfx1250.
//
// Design (MI455X, compile-only reasoning):
//  * bf16 WMMA 16x16x32, f32 accumulate, for both the recurrent GEMM
//    (h_{t-1} @ W_hh^T) and the layer-1 input GEMM (h1_t @ W_ih^T).
//  * W_hh B-tiles resident in VGPRs across the whole time loop; h state
//    ping-pongs through LDS; c state stays in accumulator-layout VGPRs.
//  * Batch rows are recurrence-independent: grid = (2 dirs) x (4 blocks of
//    64 rows); block = 512 threads = 16 wave32 waves = 2 m-halves x 8 j-blocks.
//    Each wave owns 16 hidden columns across all 4 gates -> gate fusion is
//    lane-local in the WMMA C layout (M = r + 8*laneHi, N = lane%16).
//  * Gate activations use gfx1250 native v_tanh_f32 (one TRANS op per gate,
//    sigmoid(x) = 0.5*tanh(x/2)+0.5) -- no IEEE division chains on the
//    serial critical path.
//  * Layer-0 emits h1 as bf16 [T][256][256] into d_ws (~64 MiB assumed
//    available); layer-1 streams it back with global_prefetch_b8 lookahead.
//  * One s_barrier per time step (x-stage is double-buffered in layer 0).
// ---------------------------------------------------------------------------

typedef __bf16 bf16;
typedef __attribute__((ext_vector_type(16))) __bf16 bf16x16;
typedef __attribute__((ext_vector_type(8)))  float  f32x8;

union V16 { bf16x16 v; uint4 q[2]; };
static_assert(sizeof(V16) == 32, "bf16x16 must be 32 bytes");

#define T_STEPS 512
#define BATCH   256
#define HID     128

#define WHH_TILES  (32 * 4 * 512)   // bf16 elems per packed W_hh  (NT=32,KT=4)
#define WIH1_TILES (32 * 8 * 512)   // bf16 elems per packed W_ih1 (NT=32,KT=8)

__device__ __forceinline__ float fast_rcp(float x) {
#if __has_builtin(__builtin_amdgcn_rcpf)
    return __builtin_amdgcn_rcpf(x);      // v_rcp_f32, no div fixup
#else
    return 1.0f / x;
#endif
}

__device__ __forceinline__ float fast_tanh(float x) {
#if __has_builtin(__builtin_amdgcn_tanhf)
    return __builtin_amdgcn_tanhf(x);     // gfx1250 native v_tanh_f32
#else
    float e = __expf(-2.0f * fabsf(x));
    float t = (1.0f - e) * fast_rcp(1.0f + e);
    return copysignf(t, x);
#endif
}

__device__ __forceinline__ float sigm(float x) {
    return fmaf(0.5f, fast_tanh(0.5f * x), 0.5f);
}

// Pack a row-major f32 weight matrix W[N=4H rows][Kdim cols] into bf16 WMMA
// B-operand tiles (K32 x N16). Register layout (ISA 7.12.2 dense-B style):
//   lane L: N = nt*16 + L%16;  K = kt*32 + (r>=8 ? 16 : 0) + (L>=16 ? 8 : 0) + r%8, r = v*2 + h
// Storage: tile-major, then lane*16 + v*2 + h  -> in-kernel load is two
// contiguous b128 loads per lane.
__global__ void pack_b_tiles(const float* __restrict__ W, bf16* __restrict__ out,
                             int Kdim, int NT, int KT)
{
    int idx = blockIdx.x * blockDim.x + threadIdx.x;
    int total = NT * KT * 512;
    if (idx >= total) return;
    int e    = idx & 511;
    int tile = idx >> 9;
    int kt = tile % KT, nt = tile / KT;
    int lane = e >> 4, r = e & 15;
    int v = r >> 1, h = r & 1;
    int K = kt * 32 + ((r >= 8) ? 16 : 0) + ((lane >= 16) ? 8 : 0) + (r & 7);
    int N = nt * 16 + (lane & 15);
    out[idx] = (bf16)W[(size_t)N * Kdim + K];
}

// ---------------------------------------------------------------------------
// Layer 0: input size 3 (x-projection done with 3 VALU FMAs per element),
// recurrent GEMM via WMMA. Writes h1 (bf16) for layer 1.
// grid (2 dirs, 4 row-blocks), block 512.
// ---------------------------------------------------------------------------
__global__ __launch_bounds__(512)
void lstm_l0(const float* __restrict__ x,       // [256][512][3]
             const bf16* __restrict__ whhp,     // 2 packed W_hh (dir-major)
             const float* __restrict__ wihF, const float* __restrict__ wihB, // [512][3]
             const float* __restrict__ bF,   const float* __restrict__ bB,   // [512]
             bf16* __restrict__ h1)             // [512][256][256]
{
    extern __shared__ char smem[];
    bf16*  hbuf = (bf16*)smem;                                   // 2 * 64*128
    float* xst  = (float*)(smem + 2 * 64 * 128 * sizeof(bf16));  // 2 * 64*3

    const int dir  = blockIdx.x;
    const int row0 = blockIdx.y * 64;
    const int wave = threadIdx.x >> 5;
    const int lane = threadIdx.x & 31;
    const int jb = wave & 7, mh = wave >> 3;
    const int j0 = jb * 16;
    const int laneN = lane & 15, laneHi = lane >> 4;

    const bf16*  wq = whhp + (size_t)dir * WHH_TILES;
    const float* wx = dir ? wihB : wihF;
    const float* bs = dir ? bB : bF;

    // Recurrent weights: 16 B-tiles resident in VGPRs for the whole loop.
    bf16x16 wb[4][4];
    #pragma unroll
    for (int g = 0; g < 4; ++g)
        #pragma unroll
        for (int kt = 0; kt < 4; ++kt) {
            const bf16* p = wq + (size_t)((g * 8 + jb) * 4 + kt) * 512 + lane * 16;
            V16 t; t.q[0] = *(const uint4*)p; t.q[1] = *(const uint4*)(p + 8);
            wb[g][kt] = t.v;
        }

    // Per-lane input-projection weights and bias (gate columns this lane owns).
    float wxl[4][3], bl[4];
    #pragma unroll
    for (int g = 0; g < 4; ++g) {
        int n = g * 128 + j0 + laneN;
        bl[g] = bs[n];
        wxl[g][0] = wx[n * 3 + 0];
        wxl[g][1] = wx[n * 3 + 1];
        wxl[g][2] = wx[n * 3 + 2];
    }

    for (int i = threadIdx.x; i < 2 * 64 * 128; i += 512) hbuf[i] = (bf16)0.0f;

    // Stage x for t = 0 into buffer 0 (staging for t+1 overlaps compute later;
    // the single end-of-step barrier covers both the h ping-pong and x stage).
    const int sr = threadIdx.x / 3, sk = threadIdx.x - sr * 3;  // valid when < 192
    if (threadIdx.x < 192) {
        int tt0 = dir ? (T_STEPS - 1) : 0;
        xst[threadIdx.x] = x[(size_t)(row0 + sr) * (T_STEPS * 3) + tt0 * 3 + sk];
    }

    f32x8 zero8 = {0.f, 0.f, 0.f, 0.f, 0.f, 0.f, 0.f, 0.f};
    f32x8 cst[2] = {zero8, zero8};
    __syncthreads();

    int cur = 0;
    for (int t = 0; t < T_STEPS; ++t) {
        const int tt = dir ? (T_STEPS - 1 - t) : t;

        // Stage x[t+1] into the other x buffer (no barrier needed here: the
        // end-of-previous-step barrier guarantees its readers are done).
        if (threadIdx.x < 192 && t + 1 < T_STEPS) {
            int ttn = dir ? (tt - 1) : (tt + 1);
            const float* px = &x[(size_t)(row0 + sr) * (T_STEPS * 3) + ttn * 3 + sk];
            xst[((t + 1) & 1) * 192 + threadIdx.x] = *px;
            __builtin_prefetch(px + (dir ? -3 : 3), 0, 0);   // t+2 lookahead
        }
        const float* xcur = xst + (t & 1) * 192;

        const bf16* hc = hbuf + cur * (64 * 128);
        bf16*       hn = hbuf + (cur ^ 1) * (64 * 128);

        #pragma unroll
        for (int mi = 0; mi < 2; ++mi) {
            const int mrow = (mh * 2 + mi) * 16;

            // Accumulators start as bias + x-projection (in=3).
            f32x8 acc[4];
            #pragma unroll
            for (int r = 0; r < 8; ++r) {
                const int Ml = mrow + r + laneHi * 8;
                float x0 = xcur[Ml * 3 + 0], x1 = xcur[Ml * 3 + 1], x2 = xcur[Ml * 3 + 2];
                #pragma unroll
                for (int g = 0; g < 4; ++g)
                    acc[g][r] = bl[g] + x0 * wxl[g][0] + x1 * wxl[g][1] + x2 * wxl[g][2];
            }

            // Recurrent GEMM: A = h_{t-1} 16x32 tiles from LDS (two 16B K-runs
            // per lane, ISA A-layout), B = resident W_hh tiles.
            const int arow = mrow + laneN;
            #pragma unroll
            for (int kt = 0; kt < 4; ++kt) {
                V16 a;
                const bf16* pa = hc + arow * 128 + kt * 32 + laneHi * 8;
                a.q[0] = *(const uint4*)pa;
                a.q[1] = *(const uint4*)(pa + 16);
                #pragma unroll
                for (int g = 0; g < 4; ++g)
                    acc[g] = __builtin_amdgcn_wmma_f32_16x16x32_bf16(
                        false, a.v, false, wb[g][kt], (short)0, acc[g], false, false);
            }

            // Gate fusion: i,f,g,o are lane-local in C-layout. One native
            // v_tanh_f32 per gate value.
            f32x8 cn, hv;
            #pragma unroll
            for (int r = 0; r < 8; ++r) {
                float ig = sigm(acc[0][r]);
                float fg = sigm(acc[1][r]);
                float gg = fast_tanh(acc[2][r]);
                float og = sigm(acc[3][r]);
                float cc = fg * cst[mi][r] + ig * gg;
                cn[r] = cc;
                hv[r] = og * fast_tanh(cc);
            }
            cst[mi] = cn;

            const size_t ob = (size_t)tt * (BATCH * 256) + (size_t)(dir ? 128 : 0) + j0 + laneN;
            #pragma unroll
            for (int r = 0; r < 8; ++r) {
                const int Ml = mrow + r + laneHi * 8;
                bf16 hb = (bf16)hv[r];
                hn[Ml * 128 + j0 + laneN] = hb;                 // next-step A source
                h1[ob + (size_t)(row0 + Ml) * 256] = hb;        // layer-1 input
            }
        }
        __syncthreads();
        cur ^= 1;
    }
}

// ---------------------------------------------------------------------------
// Layer 1: input GEMM (K=256, from h1 in global) folded into each recurrence
// step + recurrent GEMM (K=128, from LDS). Only h2[T-1] is needed downstream.
// ---------------------------------------------------------------------------
__global__ __launch_bounds__(512)
void lstm_l1(const bf16* __restrict__ h1,       // [512][256][256]
             const bf16* __restrict__ whhp,     // 2 packed W_hh (dir-major)
             const bf16* __restrict__ wih1p,    // 2 packed W_ih (dir-major)
             const float* __restrict__ bF, const float* __restrict__ bB,
             float* __restrict__ h2last)        // [256][256] f32
{
    extern __shared__ char smem[];
    bf16* hbuf = (bf16*)smem;  // 2 * 64*128

    const int dir  = blockIdx.x;
    const int row0 = blockIdx.y * 64;
    const int wave = threadIdx.x >> 5;
    const int lane = threadIdx.x & 31;
    const int jb = wave & 7, mh = wave >> 3;
    const int j0 = jb * 16;
    const int laneN = lane & 15, laneHi = lane >> 4;

    const bf16* wq  = whhp  + (size_t)dir * WHH_TILES;
    const bf16* wiq = wih1p + (size_t)dir * WIH1_TILES;
    const float* bs = dir ? bB : bF;

    bf16x16 wb[4][4];
    #pragma unroll
    for (int g = 0; g < 4; ++g)
        #pragma unroll
        for (int kt = 0; kt < 4; ++kt) {
            const bf16* p = wq + (size_t)((g * 8 + jb) * 4 + kt) * 512 + lane * 16;
            V16 t; t.q[0] = *(const uint4*)p; t.q[1] = *(const uint4*)(p + 8);
            wb[g][kt] = t.v;
        }

    float bl[4];
    #pragma unroll
    for (int g = 0; g < 4; ++g) bl[g] = bs[g * 128 + j0 + laneN];

    for (int i = threadIdx.x; i < 2 * 64 * 128; i += 512) hbuf[i] = (bf16)0.0f;

    f32x8 zero8 = {0.f, 0.f, 0.f, 0.f, 0.f, 0.f, 0.f, 0.f};
    f32x8 cst[2] = {zero8, zero8};
    __syncthreads();

    int cur = 0;
    for (int t = 0; t < T_STEPS; ++t) {
        const int tt = dir ? (T_STEPS - 1 - t) : t;
        const bf16* h1t = h1 + (size_t)tt * (BATCH * 256);

        {   // look-ahead prefetch of the next time slice of this row block
            int ttn = dir ? (tt > 0 ? tt - 1 : tt) : (tt < T_STEPS - 1 ? tt + 1 : tt);
            const bf16* pn = h1 + (size_t)ttn * (BATCH * 256)
                                + (size_t)row0 * 256 + (size_t)threadIdx.x * 32;
            __builtin_prefetch(pn, 0, 0);
        }

        const bf16* hc = hbuf + cur * (64 * 128);
        bf16*       hn = hbuf + (cur ^ 1) * (64 * 128);

        #pragma unroll
        for (int mi = 0; mi < 2; ++mi) {
            const int mrow = (mh * 2 + mi) * 16;

            f32x8 acc[4];
            #pragma unroll
            for (int g = 0; g < 4; ++g) {
                f32x8 tb;
                #pragma unroll
                for (int r = 0; r < 8; ++r) tb[r] = bl[g];
                acc[g] = tb;
            }

            // Recurrent part: K = 128 from LDS.
            const int arow = mrow + laneN;
            #pragma unroll
            for (int kt = 0; kt < 4; ++kt) {
                V16 a;
                const bf16* pa = hc + arow * 128 + kt * 32 + laneHi * 8;
                a.q[0] = *(const uint4*)pa;
                a.q[1] = *(const uint4*)(pa + 16);
                #pragma unroll
                for (int g = 0; g < 4; ++g)
                    acc[g] = __builtin_amdgcn_wmma_f32_16x16x32_bf16(
                        false, a.v, false, wb[g][kt], (short)0, acc[g], false, false);
            }

            // Input part: K = 256 from global h1[tt] (L2-resident slice),
            // W_ih tiles streamed from packed ws (WGP$-resident).
            const int arowg = row0 + mrow + laneN;
            #pragma unroll
            for (int kt2 = 0; kt2 < 8; ++kt2) {
                V16 ai;
                const bf16* pa = h1t + (size_t)arowg * 256 + kt2 * 32 + laneHi * 8;
                ai.q[0] = *(const uint4*)pa;
                ai.q[1] = *(const uint4*)(pa + 16);
                #pragma unroll
                for (int g = 0; g < 4; ++g) {
                    const bf16* pw = wiq + (size_t)((g * 8 + jb) * 8 + kt2) * 512 + lane * 16;
                    V16 w; w.q[0] = *(const uint4*)pw; w.q[1] = *(const uint4*)(pw + 8);
                    acc[g] = __builtin_amdgcn_wmma_f32_16x16x32_bf16(
                        false, ai.v, false, w.v, (short)0, acc[g], false, false);
                }
            }

            f32x8 cn, hv;
            #pragma unroll
            for (int r = 0; r < 8; ++r) {
                float ig = sigm(acc[0][r]);
                float fg = sigm(acc[1][r]);
                float gg = fast_tanh(acc[2][r]);
                float og = sigm(acc[3][r]);
                float cc = fg * cst[mi][r] + ig * gg;
                cn[r] = cc;
                hv[r] = og * fast_tanh(cc);
            }
            cst[mi] = cn;

            #pragma unroll
            for (int r = 0; r < 8; ++r) {
                const int Ml = mrow + r + laneHi * 8;
                hn[Ml * 128 + j0 + laneN] = (bf16)hv[r];
            }
            if (tt == T_STEPS - 1) {   // only h2[-1] feeds the FC head
                #pragma unroll
                for (int r = 0; r < 8; ++r) {
                    const int Ml = mrow + r + laneHi * 8;
                    h2last[(size_t)(row0 + Ml) * 256 + (dir ? 128 : 0) + j0 + laneN] = hv[r];
                }
            }
        }
        __syncthreads();
        cur ^= 1;
    }
}

// Tiny FC head: [256,256] x [256,6]^T + b. 1536 FMAs/thread — negligible.
__global__ void fc_head(const float* __restrict__ h2, const float* __restrict__ Wfc,
                        const float* __restrict__ bfc, float* __restrict__ out)
{
    int b = blockIdx.x * blockDim.x + threadIdx.x;
    if (b >= BATCH) return;
    const float* hr = h2 + (size_t)b * 256;
    #pragma unroll
    for (int o = 0; o < 6; ++o) {
        const float* wr = Wfc + o * 256;
        float s = bfc[o];
        for (int j = 0; j < 256; ++j) s += hr[j] * wr[j];
        out[b * 6 + o] = s;
    }
}

extern "C" void kernel_launch(void* const* d_in, const int* in_sizes, int n_in,
                              void* d_out, int out_size, void* d_ws, size_t ws_size,
                              hipStream_t stream)
{
    const float* x     = (const float*)d_in[0];
    const float* Wih0f = (const float*)d_in[1];
    const float* Whh0f = (const float*)d_in[2];
    const float* b0f   = (const float*)d_in[3];
    const float* Wih0b = (const float*)d_in[4];
    const float* Whh0b = (const float*)d_in[5];
    const float* b0b   = (const float*)d_in[6];
    const float* Wih1f = (const float*)d_in[7];
    const float* Whh1f = (const float*)d_in[8];
    const float* b1f   = (const float*)d_in[9];
    const float* Wih1b = (const float*)d_in[10];
    const float* Whh1b = (const float*)d_in[11];
    const float* b1b   = (const float*)d_in[12];
    const float* Wfc   = (const float*)d_in[13];
    const float* bfc   = (const float*)d_in[14];
    float* out = (float*)d_out;

    // Workspace layout (~65.3 MiB): h1(bf16) | 4x W_hh packed | 2x W_ih1 packed | h2last(f32)
    char* ws = (char*)d_ws;
    bf16*  h1    = (bf16*)ws;
    bf16*  whhp  = (bf16*)(ws + (size_t)T_STEPS * BATCH * 256 * sizeof(bf16));
    bf16*  wih1p = whhp + 4 * (size_t)WHH_TILES;
    float* h2    = (float*)(wih1p + 2 * (size_t)WIH1_TILES);

    const int thr = 256;
    // Pack recurrent weights (K=128, NT=32, KT=4): order l0f, l0b, l1f, l1b.
    pack_b_tiles<<<(WHH_TILES + thr - 1) / thr, thr, 0, stream>>>(Whh0f, whhp + 0 * (size_t)WHH_TILES, 128, 32, 4);
    pack_b_tiles<<<(WHH_TILES + thr - 1) / thr, thr, 0, stream>>>(Whh0b, whhp + 1 * (size_t)WHH_TILES, 128, 32, 4);
    pack_b_tiles<<<(WHH_TILES + thr - 1) / thr, thr, 0, stream>>>(Whh1f, whhp + 2 * (size_t)WHH_TILES, 128, 32, 4);
    pack_b_tiles<<<(WHH_TILES + thr - 1) / thr, thr, 0, stream>>>(Whh1b, whhp + 3 * (size_t)WHH_TILES, 128, 32, 4);
    // Pack layer-1 input weights (K=256, NT=32, KT=8): order l1f, l1b.
    pack_b_tiles<<<(WIH1_TILES + thr - 1) / thr, thr, 0, stream>>>(Wih1f, wih1p + 0 * (size_t)WIH1_TILES, 256, 32, 8);
    pack_b_tiles<<<(WIH1_TILES + thr - 1) / thr, thr, 0, stream>>>(Wih1b, wih1p + 1 * (size_t)WIH1_TILES, 256, 32, 8);

    size_t smem0 = 2 * 64 * 128 * sizeof(bf16) + 2 * 64 * 3 * sizeof(float);
    size_t smem1 = 2 * 64 * 128 * sizeof(bf16);
    lstm_l0<<<dim3(2, 4), 512, smem0, stream>>>(x, whhp, Wih0f, Wih0b, b0f, b0b, h1);
    lstm_l1<<<dim3(2, 4), 512, smem1, stream>>>(h1, whhp + 2 * (size_t)WHH_TILES, wih1p, b1f, b1b, h2);
    fc_head<<<1, 256, 0, stream>>>(h2, Wfc, bfc, out);

    (void)in_sizes; (void)n_in; (void)out_size; (void)ws_size;
}
